// LongformerForLEDGAR_29111288333016
// MI455X (gfx1250) — compile-verified, matrix operand order checked by reading, not run on hardware
//
#include <hip/hip_runtime.h>
#include <hip/hip_bf16.h>
#include <math.h>

typedef __attribute__((ext_vector_type(16))) __bf16 v16bf;
typedef __attribute__((ext_vector_type(8)))  __bf16 v8bf;
typedef __attribute__((ext_vector_type(8)))  float  v8f;
typedef __attribute__((ext_vector_type(4)))  unsigned int v4u;
typedef __attribute__((ext_vector_type(8)))  int   v8i;
typedef __attribute__((ext_vector_type(4)))  int   v4i;

#define S_LEN   4096
#define D_MODEL 768
#define H_HEADS 12
#define NLAYER  12
#define FFN_DIM 3072
#define NCHUNK  64
#define Q_SCALE 0.125f      /* 1/sqrt(64) */
#define NEG_BIG -1.0e9f

#ifndef __has_builtin
#define __has_builtin(x) 0
#endif
#if __has_builtin(__builtin_amdgcn_tensor_load_to_lds) && \
    __has_builtin(__builtin_amdgcn_s_wait_tensorcnt)
#define HAVE_TDM 1
#else
#define HAVE_TDM 0
#endif

// ---------------------------------------------------------------------------
// WMMA helpers (CDNA5 bf16 16x16x32, fp32 accumulate)
// ---------------------------------------------------------------------------
__device__ __forceinline__ v16bf load_frag(const __bf16* p) {
  // per-lane fragment: elements K=kb..kb+7 then K=kb+16..kb+23 (kb handled by caller)
  const v8bf lo = *(const v8bf*)(p);
  const v8bf hi = *(const v8bf*)(p + 16);
  return __builtin_shufflevector(lo, hi, 0,1,2,3,4,5,6,7,8,9,10,11,12,13,14,15);
}

__device__ __forceinline__ v8f wmma_bf16(v16bf a, v16bf b, v8f c) {
  return __builtin_amdgcn_wmma_f32_16x16x32_bf16(false, a, false, b, (short)0, c,
                                                 false, false);
}

__device__ __forceinline__ float block_sum256(float x, float* red, int lane, int w) {
  #pragma unroll
  for (int off = 16; off > 0; off >>= 1) x += __shfl_xor(x, off, 32);
  if (lane == 0) red[w] = x;
  __syncthreads();
  float t = 0.f;
  #pragma unroll
  for (int i = 0; i < 8; ++i) t += red[i];
  __syncthreads();
  return t;
}

#if HAVE_TDM
// ---------------------------------------------------------------------------
// Tensor Data Mover: async DMA of a 2D bf16 tile (rows x 64 elts, row stride
// `stride_elts`) from global to LDS at byte offset `lds_off`, with LDS row
// padding 32dw data + 4dw pad => 144B rows (matches the 72-elt LDS stride).
// Descriptor packing per CDNA5 ISA ch.8 (D# groups 0/1).
// ---------------------------------------------------------------------------
__device__ __forceinline__ void tdm_load_tile(const __bf16* g, unsigned lds_off,
                                              unsigned rows, unsigned stride_elts) {
  const unsigned long long ga = (unsigned long long)(size_t)g;
  v4u g0 = {};
  g0[0] = 1u;                                   // count=1 (valid user descriptor)
  g0[1] = lds_off;                              // lds_addr (bytes)
  g0[2] = (unsigned)ga;                         // global_addr[31:0]
  g0[3] = (unsigned)((ga >> 32) & 0x01ffffffu)  // global_addr[56:32]
        | (2u << 30);                           // type = 2 ("image")
  const unsigned td0 = stride_elts;             // tensor_dim0 (elements)
  const unsigned td1 = rows;                    // tensor_dim1
  v8i g1 = {};
  g1[0] = (int)((1u << 16)        // data_size = 2 bytes
              | (1u << 20)        // pad_enable
              | (4u << 22)        // pad_interval: 32 DWORDs
              | (3u << 25));      // pad_amount:   4 DWORDs
  g1[1] = (int)((td0 & 0xffffu) << 16);                           // tensor_dim0 lo
  g1[2] = (int)(((td0 >> 16) & 0xffffu) | ((td1 & 0xffffu) << 16));
  g1[3] = (int)(((td1 >> 16) & 0xffffu) | (64u << 16));           // tile_dim0 = 64
  g1[4] = (int)rows;                                              // tile_dim1
  g1[5] = (int)stride_elts;                                       // dim0 stride lo
  g1[6] = 0; g1[7] = 0;
  const v4i gz = {};
#if __clang_major__ >= 23
  const v8i gz8 = {};
  __builtin_amdgcn_tensor_load_to_lds(g0, g1, gz, gz, gz8, 0);
#else
  __builtin_amdgcn_tensor_load_to_lds(g0, g1, gz, gz, 0);
#endif
}
#endif // HAVE_TDM

// ---------------------------------------------------------------------------
// Weight prep: fp32 [K][N]  ->  bf16 [N][K]  (so GEMM B-fragments load like A)
// ---------------------------------------------------------------------------
__global__ void cvt_transpose(const float* __restrict__ src, __bf16* __restrict__ dst,
                              int K, int N) {
  const size_t i = (size_t)blockIdx.x * blockDim.x + threadIdx.x;
  if (i >= (size_t)K * N) return;
  const int k = (int)(i / N), n = (int)(i % N);
  dst[(size_t)n * K + k] = (__bf16)src[i];
}

// ---------------------------------------------------------------------------
// Shared GEMM inner tile: 2x4 WMMA tiles per wave over a 64-wide K slice
// ---------------------------------------------------------------------------
__device__ __forceinline__ void gemm_compute_tile(
    const __bf16* __restrict__ at, const __bf16* __restrict__ bt,
    v8f (&acc)[2][4], int wr, int wc, int lr, int kb)
{
  #pragma unroll
  for (int kk = 0; kk < 64; kk += 32) {
    v16bf af[2], bfr[4];
    #pragma unroll
    for (int t = 0; t < 2; ++t)
      af[t] = load_frag(at + (wr + t * 16 + lr) * 72 + kk + kb);
    #pragma unroll
    for (int u = 0; u < 4; ++u)
      bfr[u] = load_frag(bt + (wc + u * 16 + lr) * 72 + kk + kb);
    #pragma unroll
    for (int t = 0; t < 2; ++t)
      #pragma unroll
      for (int u = 0; u < 4; ++u)
        acc[t][u] = wmma_bf16(af[t], bfr[u], acc[t][u]);
  }
}

// ---------------------------------------------------------------------------
// GEMM: out = act( alpha * (A[M,K](bf16) @ WT[N,K]^T + bias) )
// 256 threads = 8 waves; 128x128 C tile; wave = 32x64 (2x4 WMMA tiles).
// TDM path: wave 0 double-buffers 128x64 tiles via tensor_load_to_lds while
// all waves run WMMA; fallback: synchronous per-thread staging.
// ---------------------------------------------------------------------------
template<int ACT>  // 0 = none, 1 = exact GELU
__global__ __launch_bounds__(256) void gemm_bf16(
    const __bf16* __restrict__ A, const __bf16* __restrict__ WT,
    const float* __restrict__ bias, float alpha,
    float* __restrict__ outF, __bf16* __restrict__ outB,
    int M, int N, int K)
{
  __shared__ __bf16 a_lds[2][128 * 72];
  __shared__ __bf16 b_lds[2][128 * 72];

  const int tid  = threadIdx.x;
  const int lane = tid & 31, w = tid >> 5;
  const int bm = blockIdx.y * 128, bn = blockIdx.x * 128;
  const int wr = (w >> 1) * 32;           // wave row offset (0,32,64,96)
  const int wc = (w & 1) * 64;            // wave col offset (0,64)
  const int kb = (lane & 16) ? 8 : 0;     // per-lane K base
  const int lr = lane & 15;

  const v8f zero = {};
  v8f acc[2][4];
  #pragma unroll
  for (int t = 0; t < 2; ++t)
    #pragma unroll
    for (int u = 0; u < 4; ++u) acc[t][u] = zero;

#if HAVE_TDM
  const int nt = K >> 6;
  const unsigned ao[2] = { (unsigned)(size_t)(void*)&a_lds[0][0],
                           (unsigned)(size_t)(void*)&a_lds[1][0] };
  const unsigned bo[2] = { (unsigned)(size_t)(void*)&b_lds[0][0],
                           (unsigned)(size_t)(void*)&b_lds[1][0] };
  if (tid < 32) {       // wave 0 drives the TDM (EXEC-independent, TENSORcnt)
    tdm_load_tile(A  + (size_t)bm * K, ao[0], 128, (unsigned)K);
    tdm_load_tile(WT + (size_t)bn * K, bo[0], 128, (unsigned)K);
  }
  for (int it = 0; it < nt; ++it) {
    const int buf = it & 1;
    if (tid < 32) {
      if (it + 1 < nt) {
        const int k0n = (it + 1) << 6;
        tdm_load_tile(A  + (size_t)bm * K + k0n, ao[buf ^ 1], 128, (unsigned)K);
        tdm_load_tile(WT + (size_t)bn * K + k0n, bo[buf ^ 1], 128, (unsigned)K);
        __builtin_amdgcn_s_wait_tensorcnt(2);  // tile `it` landed; next in flight
      } else {
        __builtin_amdgcn_s_wait_tensorcnt(0);
      }
    }
    __syncthreads();
    gemm_compute_tile(a_lds[buf], b_lds[buf], acc, wr, wc, lr, kb);
    __syncthreads();
  }
#else
  const int srow = tid >> 1;              // 0..127
  const int soff = (tid & 1) * 32;        // 0 or 32
  for (int k0 = 0; k0 < K; k0 += 64) {
    const int buf = (k0 >> 6) & 1;
    const uint4* ga = (const uint4*)(A  + (size_t)(bm + srow) * K + k0 + soff);
    const uint4* gb = (const uint4*)(WT + (size_t)(bn + srow) * K + k0 + soff);
    uint4* la = (uint4*)(&a_lds[buf][0] + srow * 72 + soff);
    uint4* lb = (uint4*)(&b_lds[buf][0] + srow * 72 + soff);
    #pragma unroll
    for (int i = 0; i < 4; ++i) { la[i] = ga[i]; lb[i] = gb[i]; }
    if (k0 + 64 < K) {                    // prefetch next K tile
      __builtin_prefetch(A  + (size_t)(bm + srow) * K + k0 + 64 + soff, 0, 1);
      __builtin_prefetch(WT + (size_t)(bn + srow) * K + k0 + 64 + soff, 0, 1);
    }
    __syncthreads();
    gemm_compute_tile(a_lds[buf], b_lds[buf], acc, wr, wc, lr, kb);
    __syncthreads();
  }
#endif

  const int rsel = (lane & 16) ? 8 : 0;
  #pragma unroll
  for (int t = 0; t < 2; ++t) {
    #pragma unroll
    for (int u = 0; u < 4; ++u) {
      const int col = bn + wc + u * 16 + lr;
      const float bv = bias ? bias[col] : 0.f;
      #pragma unroll
      for (int j = 0; j < 8; ++j) {
        const int row = bm + wr + t * 16 + j + rsel;
        float v = (acc[t][u][j] + bv) * alpha;
        if (ACT == 1) v = 0.5f * v * (1.f + erff(v * 0.70710678118f));
        const size_t o = (size_t)row * N + col;
        if (outF) outF[o] = v;
        if (outB) outB[o] = (__bf16)v;
      }
    }
  }
}

// ---------------------------------------------------------------------------
// Windowed attention: one block per (chunk c, head h)
// scores = q(64x64) @ k3(192x64)^T  + global column sg;  softmax;  ctx = p @ v3 + pg*v0
// ---------------------------------------------------------------------------
__global__ __launch_bounds__(256) void attn_window(
    const __bf16* __restrict__ qb, const __bf16* __restrict__ kb,
    const __bf16* __restrict__ vb, const int* __restrict__ am,
    __bf16* __restrict__ ctxb)
{
  extern __shared__ char smem[];
  __bf16* q_lds  = (__bf16*)smem;            // 64 x 72
  __bf16* k_lds  = q_lds  + 64 * 72;         // 192 x 72
  __bf16* vt_lds = k_lds  + 192 * 72;        // 64 x 200  (vt[d][j])
  __bf16* p_lds  = vt_lds + 64 * 200;        // 64 x 200  (probs bf16)
  __bf16* k0_lds = p_lds  + 64 * 200;        // 64
  __bf16* v0_lds = k0_lds + 64;              // 64
  float*  s_lds  = (float*)(v0_lds + 64);    // 64 x 200  (scores fp32, col192=global)

  const int tid = threadIdx.x, lane = tid & 31, w = tid >> 5;
  const int c = blockIdx.x, h = blockIdx.y;
  const int hc  = h * 64;
  const int q0  = c * 64;
  const int kp0 = (c - 1) * 64;

  { // q tile
    const int r = tid >> 2, seg = (tid & 3) * 16;
    const uint4* g = (const uint4*)(qb + (size_t)(q0 + r) * D_MODEL + hc + seg);
    uint4* l = (uint4*)(q_lds + r * 72 + seg);
    l[0] = g[0]; l[1] = g[1];
  }
  #pragma unroll
  for (int i = 0; i < 6; ++i) { // k3 tile (192x64), zero-padded
    const int chunk = tid + i * 256;
    const int j = chunk >> 3, seg = (chunk & 7) * 8;
    const int p = kp0 + j;
    uint4 val = make_uint4(0u, 0u, 0u, 0u);
    if (p >= 0 && p < S_LEN)
      val = *(const uint4*)(kb + (size_t)p * D_MODEL + hc + seg);
    *(uint4*)(k_lds + j * 72 + seg) = val;
  }
  #pragma unroll
  for (int i = 0; i < 48; ++i) { // v3^T tile
    const int e = tid + i * 256;
    const int j = e >> 6, d = e & 63;
    const int p = kp0 + j;
    __bf16 val = (__bf16)0.f;
    if (p >= 0 && p < S_LEN) val = vb[(size_t)p * D_MODEL + hc + d];
    vt_lds[d * 200 + j] = val;
  }
  if (tid < 64) { k0_lds[tid] = kb[hc + tid]; v0_lds[tid] = vb[hc + tid]; }
  __syncthreads();

  if (tid < 64) { // global score column: sg[row] = q_row . k(pos 0)
    float a = 0.f;
    #pragma unroll
    for (int d = 0; d < 64; ++d)
      a += (float)q_lds[tid * 72 + d] * (float)k0_lds[d];
    s_lds[tid * 200 + 192] = a;
  }

  { // windowed scores via WMMA: 48 tiles over 8 waves
    const int rt = (w & 3) * 16;
    const int cb = (w >> 2) * 96;
    const int kb2 = (lane & 16) ? 8 : 0;
    const int lr = lane & 15;
    const v8f zero = {};
    v8f sc[6];
    #pragma unroll
    for (int u = 0; u < 6; ++u) sc[u] = zero;
    #pragma unroll
    for (int kk = 0; kk < 64; kk += 32) {
      v16bf aq = load_frag(q_lds + (rt + lr) * 72 + kk + kb2);
      #pragma unroll
      for (int u = 0; u < 6; ++u) {
        v16bf bk = load_frag(k_lds + (cb + u * 16 + lr) * 72 + kk + kb2);
        sc[u] = wmma_bf16(aq, bk, sc[u]);
      }
    }
    const int rsel = (lane & 16) ? 8 : 0;
    #pragma unroll
    for (int u = 0; u < 6; ++u) {
      const int col = cb + u * 16 + lr;
      const int p = kp0 + col;
      const bool keyok = (p >= 1) && (p < S_LEN) && (am[p] > 0);
      #pragma unroll
      for (int j = 0; j < 8; ++j) {
        const int row = rt + j + rsel;
        const bool band = (col >= row) && (col <= row + 128);
        s_lds[row * 200 + col] = (band && keyok) ? sc[u][j] : NEG_BIG;
      }
    }
  }
  __syncthreads();

  for (int r = w * 8; r < w * 8 + 8; ++r) { // softmax over 193 entries / row
    float m = NEG_BIG;
    for (int col = lane; col < 193; col += 32) m = fmaxf(m, s_lds[r * 200 + col]);
    #pragma unroll
    for (int off = 16; off > 0; off >>= 1) m = fmaxf(m, __shfl_xor(m, off, 32));
    float sum = 0.f;
    for (int col = lane; col < 193; col += 32) {
      float e = __expf(s_lds[r * 200 + col] - m);
      s_lds[r * 200 + col] = e;
      sum += e;
    }
    #pragma unroll
    for (int off = 16; off > 0; off >>= 1) sum += __shfl_xor(sum, off, 32);
    const float inv = 1.f / sum;
    for (int col = lane; col < 193; col += 32) {
      const float v = s_lds[r * 200 + col] * inv;
      if (col < 192) p_lds[r * 200 + col] = (__bf16)v;
      else           s_lds[r * 200 + 192] = v;   // pg
    }
  }
  __syncthreads();

  { // ctx = probs(64x192) @ v3(192x64) + pg * v0
    const int rt  = (w >> 1) * 16;
    const int cts = (w & 1) * 2;
    const int kb2 = (lane & 16) ? 8 : 0;
    const int lr = lane & 15;
    const int rsel = (lane & 16) ? 8 : 0;
    const v8f zero = {};
    #pragma unroll
    for (int ci = 0; ci < 2; ++ci) {
      const int ct = (cts + ci) * 16;
      v8f acc = zero;
      #pragma unroll
      for (int kk = 0; kk < 192; kk += 32) {
        v16bf ap = load_frag(p_lds  + (rt + lr) * 200 + kk + kb2);
        v16bf bv = load_frag(vt_lds + (ct + lr) * 200 + kk + kb2);
        acc = wmma_bf16(ap, bv, acc);
      }
      const int d = ct + lr;
      const float v0 = (float)v0_lds[d];
      #pragma unroll
      for (int j = 0; j < 8; ++j) {
        const int row = rt + j + rsel;
        const float pg = s_lds[row * 200 + 192];
        ctxb[(size_t)(q0 + row) * D_MODEL + hc + d] = (__bf16)(acc[j] + pg * v0);
      }
    }
  }
}

// ---------------------------------------------------------------------------
// Global attention for sequence position 0 (one block per head)
// ---------------------------------------------------------------------------
__global__ __launch_bounds__(256) void attn_global(
    const __bf16* __restrict__ qgb, const __bf16* __restrict__ kgb,
    const __bf16* __restrict__ vgb, const int* __restrict__ am,
    __bf16* __restrict__ ctxb)
{
  __shared__ float pbuf[S_LEN];
  __shared__ float qgl[64];
  __shared__ float red[8];
  const int tid = threadIdx.x, lane = tid & 31, w = tid >> 5;
  const int hc = blockIdx.x * 64;
  if (tid < 64) qgl[tid] = (float)qgb[hc + tid];  // qg already * SCALE
  __syncthreads();

  float m = NEG_BIG;
  for (int j = tid; j < S_LEN; j += 256) {
    float a = 0.f;
    #pragma unroll
    for (int d = 0; d < 64; ++d)
      a += qgl[d] * (float)kgb[(size_t)j * D_MODEL + hc + d];
    a = (am[j] > 0) ? a : NEG_BIG;
    pbuf[j] = a;
    m = fmaxf(m, a);
  }
  #pragma unroll
  for (int off = 16; off > 0; off >>= 1) m = fmaxf(m, __shfl_xor(m, off, 32));
  if (lane == 0) red[w] = m;
  __syncthreads();
  m = red[0];
  #pragma unroll
  for (int i = 1; i < 8; ++i) m = fmaxf(m, red[i]);
  __syncthreads();

  float sum = 0.f;
  for (int j = tid; j < S_LEN; j += 256) {
    const float e = __expf(pbuf[j] - m);
    pbuf[j] = e;
    sum += e;
  }
  sum = block_sum256(sum, red, lane, w);
  const float inv = 1.f / sum;

  for (int d = 0; d < 64; ++d) {
    float a = 0.f;
    for (int j = tid; j < S_LEN; j += 256)
      a += pbuf[j] * (float)vgb[(size_t)j * D_MODEL + hc + d];
    a = block_sum256(a, red, lane, w);
    if (tid == 0) ctxb[hc + d] = (__bf16)(a * inv);
  }
}

// ---------------------------------------------------------------------------
// LayerNorm kernels (one block per token), fp32 + bf16 outputs
// ---------------------------------------------------------------------------
__global__ __launch_bounds__(256) void residual_ln(
    const float* __restrict__ a, const float* __restrict__ b,
    const float* __restrict__ scale, const float* __restrict__ bias,
    float* __restrict__ outF, __bf16* __restrict__ outB)
{
  __shared__ float red[8];
  const int s = blockIdx.x, tid = threadIdx.x, lane = tid & 31, w = tid >> 5;
  float v[3];
  #pragma unroll
  for (int i = 0; i < 3; ++i) {
    const int d = tid + i * 256;
    v[i] = a[(size_t)s * D_MODEL + d] + (b ? b[(size_t)s * D_MODEL + d] : 0.f);
  }
  const float mean = block_sum256(v[0] + v[1] + v[2], red, lane, w) * (1.f / D_MODEL);
  float vs = 0.f;
  #pragma unroll
  for (int i = 0; i < 3; ++i) { const float t = v[i] - mean; vs += t * t; }
  const float var = block_sum256(vs, red, lane, w) * (1.f / D_MODEL);
  const float inv = rsqrtf(var + 1e-5f);
  #pragma unroll
  for (int i = 0; i < 3; ++i) {
    const int d = tid + i * 256;
    const float o = (v[i] - mean) * inv * scale[d] + bias[d];
    outF[(size_t)s * D_MODEL + d] = o;
    if (outB) outB[(size_t)s * D_MODEL + d] = (__bf16)o;
  }
}

__global__ __launch_bounds__(256) void embed_ln(
    const int* __restrict__ ids, const float* __restrict__ etok,
    const float* __restrict__ epos, const float* __restrict__ scale,
    const float* __restrict__ bias, float* __restrict__ outF,
    __bf16* __restrict__ outB)
{
  __shared__ float red[8];
  const int s = blockIdx.x, tid = threadIdx.x, lane = tid & 31, w = tid >> 5;
  const int id = ids[s];
  float v[3];
  #pragma unroll
  for (int i = 0; i < 3; ++i) {
    const int d = tid + i * 256;
    v[i] = etok[(size_t)id * D_MODEL + d] + epos[(size_t)s * D_MODEL + d];
  }
  const float mean = block_sum256(v[0] + v[1] + v[2], red, lane, w) * (1.f / D_MODEL);
  float vs = 0.f;
  #pragma unroll
  for (int i = 0; i < 3; ++i) { const float t = v[i] - mean; vs += t * t; }
  const float var = block_sum256(vs, red, lane, w) * (1.f / D_MODEL);
  const float inv = rsqrtf(var + 1e-5f);
  #pragma unroll
  for (int i = 0; i < 3; ++i) {
    const int d = tid + i * 256;
    const float o = (v[i] - mean) * inv * scale[d] + bias[d];
    outF[(size_t)s * D_MODEL + d] = o;
    outB[(size_t)s * D_MODEL + d] = (__bf16)o;
  }
}

// ---------------------------------------------------------------------------
// Classifier head: out[100] = x[0,:] @ Wc[768,100] + bc
// ---------------------------------------------------------------------------
__global__ void classifier(const float* __restrict__ x0, const float* __restrict__ Wc,
                           const float* __restrict__ bc, float* __restrict__ out) {
  const int n = threadIdx.x;
  if (n >= 100) return;
  float a = bc[n];
  for (int d = 0; d < D_MODEL; ++d) a += x0[d] * Wc[d * 100 + n];
  out[n] = a;
}

// ---------------------------------------------------------------------------
// Host driver
// ---------------------------------------------------------------------------
extern "C" void kernel_launch(void* const* d_in, const int* in_sizes, int n_in,
                              void* d_out, int out_size, void* d_ws, size_t ws_size,
                              hipStream_t stream)
{
  (void)in_sizes; (void)n_in; (void)out_size; (void)ws_size;

  const int*   ids  = (const int*)  d_in[0];
  const int*   am   = (const int*)  d_in[1];
  const float* etok = (const float*)d_in[2];
  const float* epos = (const float*)d_in[3];
  const float* lnEs = (const float*)d_in[4];
  const float* lnEb = (const float*)d_in[5];
  const float* Wq  = (const float*)d_in[6];  const float* bq  = (const float*)d_in[7];
  const float* Wk  = (const float*)d_in[8];  const float* bk  = (const float*)d_in[9];
  const float* Wv  = (const float*)d_in[10]; const float* bv  = (const float*)d_in[11];
  const float* Wqg = (const float*)d_in[12]; const float* bqg = (const float*)d_in[13];
  const float* Wkg = (const float*)d_in[14]; const float* bkg = (const float*)d_in[15];
  const float* Wvg = (const float*)d_in[16]; const float* bvg = (const float*)d_in[17];
  const float* Wo  = (const float*)d_in[18]; const float* bo  = (const float*)d_in[19];
  const float* l1s = (const float*)d_in[20]; const float* l1b = (const float*)d_in[21];
  const float* Wf1 = (const float*)d_in[22]; const float* bf1 = (const float*)d_in[23];
  const float* Wf2 = (const float*)d_in[24]; const float* bf2 = (const float*)d_in[25];
  const float* l2s = (const float*)d_in[26]; const float* l2b = (const float*)d_in[27];
  const float* Wc  = (const float*)d_in[28]; const float* bc  = (const float*)d_in[29];

  char* ws = (char*)d_ws;
  size_t off = 0;
  auto alloc = [&](size_t bytes) -> void* {
    void* p = (void*)(ws + off);
    off = (off + bytes + 255) & ~(size_t)255;
    return p;
  };

  const size_t SD  = (size_t)S_LEN * D_MODEL;
  const size_t SF  = (size_t)S_LEN * FFN_DIM;
  const size_t DD2 = (size_t)D_MODEL * D_MODEL * 2;
  const size_t DF2 = (size_t)D_MODEL * FFN_DIM * 2;

  float*  xf    = (float*) alloc(SD * 4);
  __bf16* xb    = (__bf16*)alloc(SD * 2);
  __bf16* qbuf  = (__bf16*)alloc(SD * 2);
  __bf16* kbuf  = (__bf16*)alloc(SD * 2);
  __bf16* vbuf  = (__bf16*)alloc(SD * 2);
  __bf16* qgbuf = (__bf16*)alloc(SD * 2);
  __bf16* kgbuf = (__bf16*)alloc(SD * 2);
  __bf16* vgbuf = (__bf16*)alloc(SD * 2);
  __bf16* ctxb  = (__bf16*)alloc(SD * 2);
  float*  attnf = (float*) alloc(SD * 4);
  float*  x1f   = (float*) alloc(SD * 4);
  __bf16* x1b   = (__bf16*)alloc(SD * 2);
  __bf16* hb    = (__bf16*)alloc(SF * 2);
  float*  yf    = (float*) alloc(SD * 4);
  __bf16* wqT   = (__bf16*)alloc(DD2);
  __bf16* wkT   = (__bf16*)alloc(DD2);
  __bf16* wvT   = (__bf16*)alloc(DD2);
  __bf16* wqgT  = (__bf16*)alloc(DD2);
  __bf16* wkgT  = (__bf16*)alloc(DD2);
  __bf16* wvgT  = (__bf16*)alloc(DD2);
  __bf16* woT   = (__bf16*)alloc(DD2);
  __bf16* wf1T  = (__bf16*)alloc(DF2);
  __bf16* wf2T  = (__bf16*)alloc(DF2);

  embed_ln<<<S_LEN, 256, 0, stream>>>(ids, etok, epos, lnEs, lnEb, xf, xb);

  const dim3 gDD(D_MODEL / 128, S_LEN / 128);   // N=768
  const dim3 gDF(FFN_DIM / 128, S_LEN / 128);   // N=3072
  const size_t smemAttn = (size_t)(64*72 + 192*72 + 64*200 + 64*200 + 128) * 2
                        + (size_t)(64*200) * 4;  // 139,520 B

  for (int l = 0; l < NLAYER; ++l) {
    const size_t oDD = (size_t)l * D_MODEL * D_MODEL;
    const size_t oDF = (size_t)l * D_MODEL * FFN_DIM;
    const size_t oD  = (size_t)l * D_MODEL;
    const size_t oF  = (size_t)l * FFN_DIM;
    const int gT_DD = (D_MODEL * D_MODEL + 255) / 256;
    const int gT_DF = (D_MODEL * FFN_DIM + 255) / 256;

    cvt_transpose<<<gT_DD, 256, 0, stream>>>(Wq  + oDD, wqT,  D_MODEL, D_MODEL);
    cvt_transpose<<<gT_DD, 256, 0, stream>>>(Wk  + oDD, wkT,  D_MODEL, D_MODEL);
    cvt_transpose<<<gT_DD, 256, 0, stream>>>(Wv  + oDD, wvT,  D_MODEL, D_MODEL);
    cvt_transpose<<<gT_DD, 256, 0, stream>>>(Wqg + oDD, wqgT, D_MODEL, D_MODEL);
    cvt_transpose<<<gT_DD, 256, 0, stream>>>(Wkg + oDD, wkgT, D_MODEL, D_MODEL);
    cvt_transpose<<<gT_DD, 256, 0, stream>>>(Wvg + oDD, wvgT, D_MODEL, D_MODEL);
    cvt_transpose<<<gT_DD, 256, 0, stream>>>(Wo  + oDD, woT,  D_MODEL, D_MODEL);
    cvt_transpose<<<gT_DF, 256, 0, stream>>>(Wf1 + oDF, wf1T, D_MODEL, FFN_DIM);
    cvt_transpose<<<gT_DF, 256, 0, stream>>>(Wf2 + oDF, wf2T, FFN_DIM, D_MODEL);

    gemm_bf16<0><<<gDD, 256, 0, stream>>>(xb, wqT,  bq  + oD, Q_SCALE, nullptr, qbuf,  S_LEN, D_MODEL, D_MODEL);
    gemm_bf16<0><<<gDD, 256, 0, stream>>>(xb, wkT,  bk  + oD, 1.f,     nullptr, kbuf,  S_LEN, D_MODEL, D_MODEL);
    gemm_bf16<0><<<gDD, 256, 0, stream>>>(xb, wvT,  bv  + oD, 1.f,     nullptr, vbuf,  S_LEN, D_MODEL, D_MODEL);
    gemm_bf16<0><<<gDD, 256, 0, stream>>>(xb, wqgT, bqg + oD, Q_SCALE, nullptr, qgbuf, S_LEN, D_MODEL, D_MODEL);
    gemm_bf16<0><<<gDD, 256, 0, stream>>>(xb, wkgT, bkg + oD, 1.f,     nullptr, kgbuf, S_LEN, D_MODEL, D_MODEL);
    gemm_bf16<0><<<gDD, 256, 0, stream>>>(xb, wvgT, bvg + oD, 1.f,     nullptr, vgbuf, S_LEN, D_MODEL, D_MODEL);

    attn_window<<<dim3(NCHUNK, H_HEADS), 256, smemAttn, stream>>>(qbuf, kbuf, vbuf, am, ctxb);
    attn_global<<<H_HEADS, 256, 0, stream>>>(qgbuf, kgbuf, vgbuf, am, ctxb);

    gemm_bf16<0><<<gDD, 256, 0, stream>>>(ctxb, woT, bo + oD, 1.f, attnf, nullptr, S_LEN, D_MODEL, D_MODEL);
    residual_ln<<<S_LEN, 256, 0, stream>>>(xf, attnf, l1s + oD, l1b + oD, x1f, x1b);
    gemm_bf16<1><<<gDF, 256, 0, stream>>>(x1b, wf1T, bf1 + oF, 1.f, nullptr, hb, S_LEN, FFN_DIM, D_MODEL);
    gemm_bf16<0><<<gDD, 256, 0, stream>>>(hb, wf2T, bf2 + oD, 1.f, yf, nullptr, S_LEN, D_MODEL, FFN_DIM);
    residual_ln<<<S_LEN, 256, 0, stream>>>(x1f, yf, l2s + oD, l2b + oD, xf, xb);
  }

  classifier<<<1, 128, 0, stream>>>(xf, Wc, bc, (float*)d_out);
}
